// SlotAttention_27857157882330
// MI455X (gfx1250) — matrix operand order repaired and checked by reading, not compiled, not run on hardware
//
#include <hip/hip_runtime.h>
#include <math.h>

#define LN_EPS 1e-5f
#define ATTN_EPS 1e-8f

static constexpr int BATCH = 2;
static constexpr int KSLOT = 8;
static constexpr int SDIM  = 64;
static constexpr int IDIM  = 32;
static constexpr int HDIM  = 128;
static constexpr int G0  = 80;
static constexpr int DD1 = 78;
static constexpr int DD2 = 76;
static constexpr int DD3 = 74;
static constexpr int NPOS   = DD3 * DD3 * DD3;          // 405224
static constexpr int CHUNK  = 1024;
static constexpr int SUBT   = 256;
static constexpr int NCHUNK = (NPOS + CHUNK - 1) / CHUNK; // 396
static constexpr int PARTSZ = KSLOT * IDIM + KSLOT;       // 264 floats per (block,b)
static constexpr int WPACKN = 27 * 2 * 32 * 16;           // packed bf16 elems per conv

typedef __attribute__((ext_vector_type(16))) __bf16 v16bf;
typedef __attribute__((ext_vector_type(8)))  float  v8f;

union ABf { uint4 q[2]; v16bf v; };

__device__ __forceinline__ unsigned short f2bf(float f) {
    unsigned int u = __float_as_uint(f);
    u += 0x7fffu + ((u >> 16) & 1u);       // round-to-nearest-even
    return (unsigned short)(u >> 16);
}
__device__ __forceinline__ float bf2f(unsigned short h) {
    return __uint_as_float(((unsigned int)h) << 16);
}

// ---------------------------------------------------------------- copy slots
__global__ __launch_bounds__(1024) void k_copy_slots(const float* __restrict__ src,
                                                     float* __restrict__ dst) {
    dst[threadIdx.x] = src[threadIdx.x];
}

// ------------------------------------------------- pack conv2/conv3 weights
// Packed B-fragment layout per tap: [tap][couthalf][lane][elem] bf16,
// lane L: col = L&15 (cout within half), K = (L>>4)*8 + e for e<8, 16 + (L>>4)*8 + (e-8) for e>=8  (cin).
__global__ __launch_bounds__(256) void k_pack_w(const float* __restrict__ w2,
                                                const float* __restrict__ w3,
                                                unsigned short* __restrict__ wpack) {
    int idx = blockIdx.x * 256 + threadIdx.x;
    if (idx >= 2 * WPACKN) return;
    int conv = idx / WPACKN;
    int rem  = idx % WPACKN;
    int tap  = rem >> 10;
    int hl   = (rem >> 9) & 1;
    int L    = (rem >> 4) & 31;
    int e    = rem & 15;
    int cout = hl * 16 + (L & 15);
    int cin  = (e >> 3) * 16 + (L >> 4) * 8 + (e & 7);
    const float* w = conv ? w3 : w2;               // [32][32][27] row-major
    float val = w[(cout * 32 + cin) * 27 + tap];
    wpack[idx] = f2bf(val);
}

// ------------------------------------------------------- conv1 (Cin=4) direct
// in: [2][4][80][80][80] f32  ->  out: channels-last bf16 [2][78][78][78][32]
__global__ __launch_bounds__(256) void k_conv1(const float* __restrict__ in,
                                               const float* __restrict__ w,
                                               const float* __restrict__ bias,
                                               unsigned short* __restrict__ out,
                                               int total) {
    int idx = blockIdx.x * 256 + threadIdx.x;
    if (idx >= total) return;
    int c = idx & 31;
    int p = idx >> 5;
    int x = p % DD1; p /= DD1;
    int y = p % DD1; p /= DD1;
    int z = p % DD1;
    int b = p / DD1;
    float acc = bias[c];
    for (int cin = 0; cin < 4; ++cin) {
        const float* ib = in + (((size_t)(b * 4 + cin) * G0 + z) * G0 + y) * G0 + x;
        const float* wb = w + (c * 4 + cin) * 27;
        #pragma unroll
        for (int dz = 0; dz < 3; ++dz)
            #pragma unroll
            for (int dy = 0; dy < 3; ++dy)
                #pragma unroll
                for (int dx = 0; dx < 3; ++dx)
                    acc += ib[((size_t)dz * G0 + dy) * G0 + dx] * wb[dz * 9 + dy * 3 + dx];
    }
    out[idx] = f2bf(acc > 0.f ? acc : 0.f);
}

// -------------------------------------------- conv2/conv3: implicit-GEMM WMMA
// Channels-last bf16, C=32. One wave computes a 32-positions x 32-couts tile:
// 4 WMMAs per tap sharing B fragments, weights staged in LDS per workgroup.
template <int IND, int INH, int INW, int OUTD, int OUTH, int OUTW>
__global__ __launch_bounds__(128) void k_conv_wmma(const unsigned short* __restrict__ in,
                                                   unsigned short* __restrict__ out,
                                                   const unsigned short* __restrict__ wpack,
                                                   const float* __restrict__ bias) {
    constexpr int TW = (OUTW + 31) / 32;
    constexpr int NT = BATCH * OUTD * OUTH * TW;

    __shared__ unsigned short wlds[WPACKN];
    for (int t = threadIdx.x; t < WPACKN / 8; t += 128)
        ((uint4*)wlds)[t] = ((const uint4*)wpack)[t];
    __syncthreads();

    int wid = blockIdx.x * 4 + (threadIdx.x >> 5);
    if (wid >= NT) return;                       // wave-uniform: EXEC stays full
    int lane = threadIdx.x & 31;
    int xt = wid % TW;   int t1 = wid / TW;
    int y  = t1 % OUTH;  int t2 = t1 / OUTH;
    int z  = t2 % OUTD;  int b  = t2 / OUTD;
    int x0 = xt * 32;
    int m    = lane & 15;
    int ksel = lane >> 4;

    v8f acc[2][2];
    #pragma unroll
    for (int mt = 0; mt < 2; ++mt)
        #pragma unroll
        for (int h = 0; h < 2; ++h)
            acc[mt][h] = {0.f,0.f,0.f,0.f,0.f,0.f,0.f,0.f};

    for (int dz = 0; dz < 3; ++dz)
        for (int dy = 0; dy < 3; ++dy)
            #pragma unroll
            for (int dx = 0; dx < 3; ++dx) {
                int tap = dz * 9 + dy * 3 + dx;
                // A fragments: two contiguous 16x32 bf16 blocks (positions x cin)
                const unsigned short* arow = in +
                    ((((size_t)(b * IND + z + dz) * INH + (y + dy)) * INW) +
                     (x0 + m + dx)) * 32;
                ABf a0, a1;
                a0.q[0] = *(const uint4*)(arow + ksel * 8);
                a0.q[1] = *(const uint4*)(arow + 16 + ksel * 8);
                a1.q[0] = *(const uint4*)(arow + 16 * 32 + ksel * 8);
                a1.q[1] = *(const uint4*)(arow + 16 * 32 + 16 + ksel * 8);
                // B fragments from LDS (pre-packed per-lane layout)
                const unsigned short* w0 = wlds + ((tap * 2 + 0) * 32 + lane) * 16;
                const unsigned short* w1 = wlds + ((tap * 2 + 1) * 32 + lane) * 16;
                ABf b0, b1;
                b0.q[0] = *(const uint4*)(w0); b0.q[1] = *(const uint4*)(w0 + 8);
                b1.q[0] = *(const uint4*)(w1); b1.q[1] = *(const uint4*)(w1 + 8);

                acc[0][0] = __builtin_amdgcn_wmma_f32_16x16x32_bf16(
                    false, a0.v, false, b0.v, (short)0, acc[0][0], false, false);
                acc[0][1] = __builtin_amdgcn_wmma_f32_16x16x32_bf16(
                    false, a0.v, false, b1.v, (short)0, acc[0][1], false, false);
                acc[1][0] = __builtin_amdgcn_wmma_f32_16x16x32_bf16(
                    false, a1.v, false, b0.v, (short)0, acc[1][0], false, false);
                acc[1][1] = __builtin_amdgcn_wmma_f32_16x16x32_bf16(
                    false, a1.v, false, b1.v, (short)0, acc[1][1], false, false);
            }

    // Epilogue: bias + relu + bf16 store (mask invalid x lanes)
    int n    = lane & 15;
    int msel = lane >> 4;
    size_t outbase = (((size_t)(b * OUTD + z) * OUTH + y) * OUTW);
    #pragma unroll
    for (int mt = 0; mt < 2; ++mt)
        #pragma unroll
        for (int h = 0; h < 2; ++h) {
            v8f A = acc[mt][h];
            int cout = h * 16 + n;
            float bb = bias[cout];
            #pragma unroll
            for (int r = 0; r < 8; ++r) {
                int x = x0 + mt * 16 + msel * 8 + r;
                if (x < OUTW) {
                    float v = A[r] + bb;
                    out[(outbase + x) * 32 + cout] = f2bf(v > 0.f ? v : 0.f);
                }
            }
        }
}

// --------------------------------------------- feature LayerNorm (bf16 -> f32)
__global__ __launch_bounds__(256) void k_ln_feat(const unsigned short* __restrict__ act,
                                                 const float* __restrict__ g,
                                                 const float* __restrict__ bta,
                                                 float* __restrict__ feat,
                                                 int total) {
    int p = blockIdx.x * 256 + threadIdx.x;
    if (p >= total) return;
    const unsigned short* row = act + (size_t)p * 32;
    float v[32];
    float mean = 0.f;
    #pragma unroll
    for (int c = 0; c < 32; ++c) { v[c] = bf2f(row[c]); mean += v[c]; }
    mean *= (1.f / 32.f);
    float var = 0.f;
    #pragma unroll
    for (int c = 0; c < 32; ++c) { float d = v[c] - mean; var += d * d; }
    var *= (1.f / 32.f);
    float rs = rsqrtf(var + LN_EPS);
    float* fo = feat + (size_t)p * 32;
    #pragma unroll
    for (int c = 0; c < 32; ++c) fo[c] = (v[c] - mean) * rs * g[c] + bta[c];
}

// --------------------------- q = LN(slots)@q_w.T ; qk = scale * (q @ k_w) -----
__global__ __launch_bounds__(1024) void k_qproj(const float* __restrict__ slots,
                                                const float* __restrict__ qg,
                                                const float* __restrict__ qb,
                                                const float* __restrict__ q_w,
                                                const float* __restrict__ k_w,
                                                float* __restrict__ qk) {
    __shared__ float Srow[16 * 64];
    __shared__ float Lrow[16 * 64];
    __shared__ float Q[16 * 64];
    int tid = threadIdx.x;
    int r = tid >> 6, d = tid & 63;
    Srow[tid] = slots[tid];
    __syncthreads();
    float mean = 0.f;
    for (int e = 0; e < 64; ++e) mean += Srow[r * 64 + e];
    mean *= (1.f / 64.f);
    float var = 0.f;
    for (int e = 0; e < 64; ++e) { float df = Srow[r * 64 + e] - mean; var += df * df; }
    var *= (1.f / 64.f);
    float rs = rsqrtf(var + LN_EPS);
    Lrow[tid] = (Srow[tid] - mean) * rs * qg[d] + qb[d];
    __syncthreads();
    float acc = 0.f;
    for (int e = 0; e < 64; ++e) acc += Lrow[r * 64 + e] * q_w[d * 64 + e];
    Q[tid] = acc;
    __syncthreads();
    if (tid < 16 * 32) {
        int rr = tid >> 5, c = tid & 31;
        float a = 0.f;
        for (int dd = 0; dd < 64; ++dd) a += Q[rr * 64 + dd] * k_w[dd * 32 + c];
        qk[rr * 32 + c] = 0.125f * a;    // SD^-0.5 folded in
    }
}

// ------------- streaming attention: softmax over slots + partial S/F sums ----
__global__ __launch_bounds__(256) void k_attn(const float* __restrict__ feat,
                                              const float* __restrict__ qk,
                                              float* __restrict__ attn_out,
                                              float* __restrict__ partial) {
    __shared__ float qkL[KSLOT * IDIM];
    __shared__ float featL[SUBT * 33];
    __shared__ float attnL[SUBT * 9];
    int tid = threadIdx.x;
    int cx  = blockIdx.x;
    int b   = blockIdx.y;
    qkL[tid] = qk[b * 256 + tid];
    __syncthreads();

    int i_b = tid >> 5, c_b = tid & 31;
    float accF = 0.f, accS = 0.f;

    for (int sub = 0; sub < CHUNK / SUBT; ++sub) {
        int p = cx * CHUNK + sub * SUBT + tid;
        bool valid = p < NPOS;
        float f[32];
        if (valid) {
            const float4* fp = (const float4*)(feat + ((size_t)b * NPOS + p) * 32);
            #pragma unroll
            for (int j = 0; j < 8; ++j) {
                float4 t = fp[j];
                f[4 * j] = t.x; f[4 * j + 1] = t.y; f[4 * j + 2] = t.z; f[4 * j + 3] = t.w;
            }
        } else {
            #pragma unroll
            for (int c = 0; c < 32; ++c) f[c] = 0.f;
        }
        #pragma unroll
        for (int c = 0; c < 32; ++c) featL[tid * 33 + c] = f[c];

        float d8[8];
        #pragma unroll
        for (int i = 0; i < 8; ++i) {
            float a = 0.f;
            #pragma unroll
            for (int c = 0; c < 32; ++c) a += qkL[i * 32 + c] * f[c];
            d8[i] = a;
        }
        float mx = d8[0];
        #pragma unroll
        for (int i = 1; i < 8; ++i) mx = fmaxf(mx, d8[i]);
        float sum = 0.f;
        #pragma unroll
        for (int i = 0; i < 8; ++i) { d8[i] = __expf(d8[i] - mx); sum += d8[i]; }
        float inv = 1.f / sum;
        #pragma unroll
        for (int i = 0; i < 8; ++i) {
            float a = valid ? (d8[i] * inv + ATTN_EPS) : 0.f;
            attnL[tid * 9 + i] = a;
            if (valid) attn_out[((size_t)(b * 8 + i)) * NPOS + p] = a;
        }
        __syncthreads();
        for (int p2 = 0; p2 < SUBT; ++p2) {
            float av = attnL[p2 * 9 + i_b];
            accF += av * featL[p2 * 33 + c_b];
            accS += av;
        }
        __syncthreads();
    }
    size_t pb = ((size_t)(b * NCHUNK + cx)) * PARTSZ;
    partial[pb + (i_b * 32 + c_b)] = accF;
    if (c_b == 0) partial[pb + 256 + i_b] = accS;
}

// --------- reduce partials -> updates -> GRU -> LN + MLP residual -> slots ---
__global__ __launch_bounds__(1024) void k_update(const float* __restrict__ partial,
                                                 const float* __restrict__ v_w,
                                                 const float* __restrict__ wih,
                                                 const float* __restrict__ whh,
                                                 const float* __restrict__ bih,
                                                 const float* __restrict__ bhh,
                                                 const float* __restrict__ rg,
                                                 const float* __restrict__ rb,
                                                 const float* __restrict__ rw1,
                                                 const float* __restrict__ rb1,
                                                 const float* __restrict__ rw2,
                                                 const float* __restrict__ rb2,
                                                 float* __restrict__ slots_cur,
                                                 float* __restrict__ out_slots) {
    __shared__ float Fs[16 * 32];
    __shared__ float Ss[16];
    __shared__ float U[16 * 64];
    __shared__ float Hp[16 * 64];
    __shared__ float Hh[16 * 64];
    __shared__ float Tt[16 * 64];
    __shared__ float HID[16 * 128];
    int tid = threadIdx.x;
    int r = tid >> 6, d = tid & 63;

    if (tid < 512) {
        int b = tid >> 8, i = (tid >> 5) & 7, c = tid & 31;
        float s = 0.f;
        for (int j = 0; j < NCHUNK; ++j)
            s += partial[((size_t)(b * NCHUNK + j)) * PARTSZ + i * 32 + c];
        Fs[(b * 8 + i) * 32 + c] = s;
    } else if (tid < 528) {
        int idx = tid - 512;
        int b = idx >> 3, i = idx & 7;
        float s = 0.f;
        for (int j = 0; j < NCHUNK; ++j)
            s += partial[((size_t)(b * NCHUNK + j)) * PARTSZ + 256 + i];
        Ss[b * 8 + i] = s;
    }
    __syncthreads();

    // updates = v_w @ (F / S)
    {
        float inv = 1.f / Ss[r];
        float acc = 0.f;
        for (int c = 0; c < 32; ++c) acc += v_w[d * 32 + c] * Fs[r * 32 + c];
        U[tid] = acc * inv;
        Hp[tid] = slots_cur[tid];
    }
    __syncthreads();

    // GRU (gates r,z,n)
    {
        float xg[3], hg[3];
        #pragma unroll
        for (int gI = 0; gI < 3; ++gI) {
            int row = gI * 64 + d;
            float ax = bih[row], ah = bhh[row];
            for (int e = 0; e < 64; ++e) {
                ax += U[r * 64 + e] * wih[row * 64 + e];
                ah += Hp[r * 64 + e] * whh[row * 64 + e];
            }
            xg[gI] = ax; hg[gI] = ah;
        }
        float rr = 1.f / (1.f + __expf(-(xg[0] + hg[0])));
        float zz = 1.f / (1.f + __expf(-(xg[1] + hg[1])));
        float nn = tanhf(xg[2] + rr * hg[2]);
        Hh[tid] = (1.f - zz) * nn + zz * Hp[tid];
    }
    __syncthreads();

    // LayerNorm of h
    {
        float mean = 0.f;
        for (int e = 0; e < 64; ++e) mean += Hh[r * 64 + e];
        mean *= (1.f / 64.f);
        float var = 0.f;
        for (int e = 0; e < 64; ++e) { float df = Hh[r * 64 + e] - mean; var += df * df; }
        var *= (1.f / 64.f);
        float rs = rsqrtf(var + LN_EPS);
        Tt[tid] = (Hh[tid] - mean) * rs * rg[d] + rb[d];
    }
    __syncthreads();

    // hidden = relu(T @ rw1.T + b1)   (HD = 128, two per thread)
    #pragma unroll
    for (int k = 0; k < 2; ++k) {
        int hh = k * 64 + d;
        float acc = rb1[hh];
        for (int e = 0; e < 64; ++e) acc += Tt[r * 64 + e] * rw1[hh * 64 + e];
        HID[r * 128 + hh] = acc > 0.f ? acc : 0.f;
    }
    __syncthreads();

    // res = hidden @ rw2.T + b2 ; slots = h + res
    {
        float acc = rb2[d];
        for (int e = 0; e < 128; ++e) acc += HID[r * 128 + e] * rw2[d * 128 + e];
        float val = Hh[tid] + acc;
        slots_cur[tid] = val;
        out_slots[tid] = val;
    }
}

// ---------------------------------------------------------------------------
extern "C" void kernel_launch(void* const* d_in, const int* in_sizes, int n_in,
                              void* d_out, int out_size, void* d_ws, size_t ws_size,
                              hipStream_t stream) {
    (void)in_sizes; (void)n_in; (void)out_size; (void)ws_size;
    const float* slots_in = (const float*)d_in[0];
    const float* oinputs  = (const float*)d_in[1];
    const float* conv_w1  = (const float*)d_in[2];
    const float* conv_b1  = (const float*)d_in[3];
    const float* conv_w2  = (const float*)d_in[4];
    const float* conv_b2  = (const float*)d_in[5];
    const float* conv_w3  = (const float*)d_in[6];
    const float* conv_b3  = (const float*)d_in[7];
    const float* k_w      = (const float*)d_in[8];
    const float* v_w      = (const float*)d_in[9];
    const float* q_ln_g   = (const float*)d_in[10];
    const float* q_ln_b   = (const float*)d_in[11];
    const float* q_w      = (const float*)d_in[12];
    const float* gru_wih  = (const float*)d_in[13];
    const float* gru_whh  = (const float*)d_in[14];
    const float* gru_bih  = (const float*)d_in[15];
    const float* gru_bhh  = (const float*)d_in[16];
    const float* res_ln_g = (const float*)d_in[17];
    const float* res_ln_b = (const float*)d_in[18];
    const float* res_w1   = (const float*)d_in[19];
    const float* res_b1   = (const float*)d_in[20];
    const float* res_w2   = (const float*)d_in[21];
    const float* res_b2   = (const float*)d_in[22];
    const float* feat_g   = (const float*)d_in[23];
    const float* feat_b   = (const float*)d_in[24];

    float* out_slots = (float*)d_out;
    float* attn_out  = (float*)d_out + BATCH * KSLOT * SDIM;

    // workspace layout (page-padded; conv edge over-reads land in later buffers)
    char* ws = (char*)d_ws;
    size_t off = 0;
    auto alloc = [&](size_t bytes) {
        size_t r = off;
        off += (bytes + 8191) & ~(size_t)4095;
        return r;
    };
    size_t o_act1  = alloc((size_t)BATCH * DD1 * DD1 * DD1 * 32 * 2);
    size_t o_act2  = alloc((size_t)BATCH * DD2 * DD2 * DD2 * 32 * 2);
    size_t o_act3  = alloc((size_t)BATCH * DD3 * DD3 * DD3 * 32 * 2);
    size_t o_feat  = alloc((size_t)BATCH * NPOS * 32 * 4);
    size_t o_wpack = alloc((size_t)2 * WPACKN * 2);
    size_t o_qk    = alloc((size_t)BATCH * KSLOT * IDIM * 4);
    size_t o_slots = alloc((size_t)BATCH * KSLOT * SDIM * 4);
    size_t o_part  = alloc((size_t)BATCH * NCHUNK * PARTSZ * 4);

    unsigned short* act1  = (unsigned short*)(ws + o_act1);
    unsigned short* act2  = (unsigned short*)(ws + o_act2);
    unsigned short* act3  = (unsigned short*)(ws + o_act3);
    float*          feat  = (float*)(ws + o_feat);
    unsigned short* wpack = (unsigned short*)(ws + o_wpack);
    float*          qk    = (float*)(ws + o_qk);
    float*          slots = (float*)(ws + o_slots);
    float*          part  = (float*)(ws + o_part);

    k_copy_slots<<<1, 1024, 0, stream>>>(slots_in, slots);
    k_pack_w<<<(2 * WPACKN + 255) / 256, 256, 0, stream>>>(conv_w2, conv_w3, wpack);

    int total1 = BATCH * DD1 * DD1 * DD1 * 32;
    k_conv1<<<(total1 + 255) / 256, 256, 0, stream>>>(oinputs, conv_w1, conv_b1, act1, total1);

    int ntiles2 = BATCH * DD2 * DD2 * ((DD2 + 31) / 32);
    k_conv_wmma<DD1, DD1, DD1, DD2, DD2, DD2>
        <<<(ntiles2 + 3) / 4, 128, 0, stream>>>(act1, act2, wpack, conv_b2);
    int ntiles3 = BATCH * DD3 * DD3 * ((DD3 + 31) / 32);
    k_conv_wmma<DD2, DD2, DD2, DD3, DD3, DD3>
        <<<(ntiles3 + 3) / 4, 128, 0, stream>>>(act2, act3, wpack + WPACKN, conv_b3);

    int totalp = BATCH * NPOS;
    k_ln_feat<<<(totalp + 255) / 256, 256, 0, stream>>>(act3, feat_g, feat_b, feat, totalp);

    for (int it = 0; it < 3; ++it) {
        k_qproj<<<1, 1024, 0, stream>>>(slots, q_ln_g, q_ln_b, q_w, k_w, qk);
        k_attn<<<dim3(NCHUNK, BATCH), 256, 0, stream>>>(feat, qk, attn_out, part);
        k_update<<<1, 1024, 0, stream>>>(part, v_w, gru_wih, gru_whh, gru_bih, gru_bhh,
                                         res_ln_g, res_ln_b, res_w1, res_b1, res_w2, res_b2,
                                         slots, out_slots);
    }
}
